// RUSAwareGatingNetworkWithAttention_15187004358767
// MI455X (gfx1250) — compile-verified, matrix operand order checked by reading, not run on hardware
//
#include <hip/hip_runtime.h>
#include <hip/hip_bf16.h>
#include <cstddef>
#include <cstdint>

// ---------------------------------------------------------------------------
// RUS-aware gating network, MI455X (gfx1250, wave32, WMMA) implementation.
//
//   prep:   tp_w -> bf16, mlp_w1[:, :256] -> bf16, c0/c1 = q_w^T k_w[:,c],
//           wU/waR/waS/wc = folded GRU input-projection vectors.
//   gemmA:  per 64-token block: processed = relu(tok@tp_w^T+b) via
//           v_wmma_f32_16x16x32_bf16 (K=768) with double-buffered LDS tiles;
//           the B tile (bf16 tp_w) is fetched with GLOBAL_LOAD_ASYNC_TO_LDS
//           DMA (ASYNCcnt) so tile k+1 streams while tile k runs on WMMA.
//           Epilogue: qk0/qk1 dots, 3-way softmax -> (aR,aS), and
//           mlpA = processed@mlp_w1[:, :256]^T + b1 via a second WMMA GEMM.
//           `processed` never touches HBM.
//   gru:    32 blocks (one per (b,m) sequence), 2048 sequential steps,
//           f32 matvecs with h/hid in LDS; writes logits.
// ---------------------------------------------------------------------------

typedef __attribute__((ext_vector_type(16))) __bf16 v16bf;
typedef __attribute__((ext_vector_type(8)))  __bf16 v8bf;
typedef __attribute__((ext_vector_type(8)))  float  v8f;

// Workspace layout (bytes, 256-aligned)
static constexpr size_t OFF_TPBF = 0;              // 256*768 bf16   (393216 B)
static constexpr size_t OFF_W1A  = 393216;         // 192*256 bf16   ( 98304 B)
static constexpr size_t OFF_C0   = 491520;         // 256 f32
static constexpr size_t OFF_C1   = 492544;         // 256 f32
static constexpr size_t OFF_D    = 493568;         // 2 f32 (qk const terms)
static constexpr size_t OFF_WU   = 493824;         // 384 f32
static constexpr size_t OFF_WAR  = 495360;         // 384 f32
static constexpr size_t OFF_WAS  = 496896;         // 384 f32
static constexpr size_t OFF_WC   = 498432;         // 384 f32
static constexpr size_t OFF_ARS  = 499968;         // 65536*2 f32    (524288 B)
static constexpr size_t OFF_MLPA = 1024256;        // 65536*192 f32  (50331648 B)

// LDS byte offset of a __shared__ object (addrspace(3) pointer value IS the
// workgroup-LDS offset, which is what GLOBAL_LOAD_ASYNC_TO_LDS's VDST wants).
#define LDS_OFF(p) ((unsigned)(uintptr_t)(__attribute__((address_space(3))) const void*)(const void*)(p))

// Load a 16x32 bf16 WMMA operand fragment (A-layout; B uses the mirrored
// mapping with n = lane&15, same indexing on an [n][k] buffer).
// ISA 7.12.2: lanes 0-15 hold rows / K {0..7,16..23}; lanes 16-31 hold
// K {8..15,24..31}. Two b128 loads per lane.
__device__ __forceinline__ v16bf frag_ld(const __bf16* base, int stride, int lane) {
  const int r  = lane & 15;
  const int kb = (lane >> 4) << 3;
  const __bf16* p = base + r * stride + kb;
  v8bf lo = *reinterpret_cast<const v8bf*>(p);       // K = kb .. kb+7
  v8bf hi = *reinterpret_cast<const v8bf*>(p + 16);  // K = kb+16 .. kb+23
  return __builtin_shufflevector(lo, hi, 0,1,2,3,4,5,6,7,8,9,10,11,12,13,14,15);
}

// ---------------------------------------------------------------------------
// Kernel 0: weight prep (bf16 conversion + algebraic folding)
// ---------------------------------------------------------------------------
__global__ __launch_bounds__(256) void prep_kernel(
    const float* __restrict__ tp_w, const float* __restrict__ mlp_w1,
    const float* __restrict__ q_w,  const float* __restrict__ q_b,
    const float* __restrict__ k_w,  const float* __restrict__ v_w,
    const float* __restrict__ v_b,  const float* __restrict__ gru_w_ih,
    const float* __restrict__ gru_b_ih,
    __bf16* __restrict__ tp_bf, __bf16* __restrict__ w1a_bf,
    float* __restrict__ c0, float* __restrict__ c1, float* __restrict__ dconst,
    float* __restrict__ wU, float* __restrict__ waR, float* __restrict__ waS,
    float* __restrict__ wc)
{
  const int gtid = blockIdx.x * 256 + threadIdx.x;
  const int nth  = gridDim.x * 256;

  for (int i = gtid; i < 256 * 768; i += nth) tp_bf[i] = (__bf16)tp_w[i];

  for (int i = gtid; i < 192 * 256; i += nth) {
    const int r = i >> 8, c = i & 255;
    w1a_bf[i] = (__bf16)mlp_w1[r * 384 + c];
  }

  for (int p = gtid; p < 256; p += nth) {
    float s0 = 0.f, s1 = 0.f;
    for (int k = 0; k < 64; ++k) {
      const float w = q_w[k * 256 + p];
      s0 += w * k_w[k * 2 + 0];
      s1 += w * k_w[k * 2 + 1];
    }
    c0[p] = s0; c1[p] = s1;
  }

  for (int i = gtid; i < 384; i += nth) {
    const float* row = gru_w_ih + (size_t)i * 65;
    float sR = 0.f, sS = 0.f, sc = 0.f;
    for (int v = 0; v < 64; ++v) {
      const float w = row[1 + v];
      sR += w * v_w[v * 2 + 0];
      sS += w * v_w[v * 2 + 1];
      sc += w * v_b[v];
    }
    wU[i]  = row[0];
    waR[i] = sR;
    waS[i] = sS;
    wc[i]  = sc + gru_b_ih[i];
  }

  if (gtid == 0) {
    float d0 = 0.f, d1 = 0.f;
    for (int k = 0; k < 64; ++k) {
      d0 += q_b[k] * k_w[k * 2 + 0];
      d1 += q_b[k] * k_w[k * 2 + 1];
    }
    dconst[0] = d0; dconst[1] = d1;
  }
}

// ---------------------------------------------------------------------------
// Kernel 1: fused WMMA GEMM (tok@tp_w^T, relu) + attention + mlpA GEMM
// 1024 blocks x 256 threads (8 waves). 64 tokens per block.
// Double-buffered LDS; B tile via async DMA to LDS.
// ---------------------------------------------------------------------------
__global__ __launch_bounds__(256) void fused_gemm_attn_kernel(
    const float* __restrict__ tok, const float* __restrict__ R,
    const float* __restrict__ S,   const float* __restrict__ tp_b,
    const float* __restrict__ mlp_b1,
    const __bf16* __restrict__ tp_bf, const __bf16* __restrict__ w1a_bf,
    const float* __restrict__ c0, const float* __restrict__ c1,
    const float* __restrict__ dconst,
    float* __restrict__ aRS, float* __restrict__ mlpA)
{
  __shared__ __bf16 At[2][64 * 32];   // A tiles (rows x k), bf16, double-buffered
  __shared__ __bf16 Bt[2][256 * 32];  // tp_w tiles, [n][k], double-buffered
  __shared__ __bf16 Pb[64 * 256];     // relu(processed) in bf16
  __shared__ float  red[64 * 4 * 2];

  const int tid  = threadIdx.x;
  const int lane = tid & 31;
  const int wave = tid >> 5;
  const int band = wave & 3;          // 16-row band within the 64-row tile
  const int half = wave >> 2;         // 0/1 -> column halves
  const int rowbase = blockIdx.x * 64;

  const unsigned long long tp_base = (unsigned long long)tp_bf;
  const unsigned ldsB0 = LDS_OFF(&Bt[0][0]);
  const unsigned ldsB1 = LDS_OFF(&Bt[1][0]);

  const int ar = tid >> 2;            // A staging: row 0..63
  const int ac = (tid & 3) * 8;       // A staging: col start {0,8,16,24}

  // stage tile `kc` into buffer `buf`:
  //  - B tile: 4x 16B GLOBAL_LOAD_ASYNC_TO_LDS_B128 DMAs per thread
  //  - A tile: f32 loads -> cvt bf16 -> one 16B LDS store per thread
  auto stage_tile = [&](int kc, int buf) {
    const unsigned lbase = (buf ? ldsB1 : ldsB0) + (unsigned)tid * 64u;
    const unsigned gbase = (unsigned)tid * 1536u + (unsigned)kc * 64u;
    #pragma unroll
    for (int j = 0; j < 4; ++j) {
      const unsigned l = lbase + (unsigned)j * 16u;
      const unsigned g = gbase + (unsigned)j * 16u;
      asm volatile("global_load_async_to_lds_b128 %0, %1, %2"
                   :: "v"(l), "v"(g), "s"(tp_base) : "memory");
    }
    const float* gp = tok + (size_t)(rowbase + ar) * 768 + kc * 32 + ac;
    float4 f0 = *reinterpret_cast<const float4*>(gp);
    float4 f1 = *reinterpret_cast<const float4*>(gp + 4);
    __bf16* apx = &At[buf][0] + ar * 32 + ac;
    apx[0] = (__bf16)f0.x; apx[1] = (__bf16)f0.y;
    apx[2] = (__bf16)f0.z; apx[3] = (__bf16)f0.w;
    apx[4] = (__bf16)f1.x; apx[5] = (__bf16)f1.y;
    apx[6] = (__bf16)f1.z; apx[7] = (__bf16)f1.w;
  };

  v8f acc[8] = {};

  stage_tile(0, 0);                   // prologue
  for (int kc = 0; kc < 24; ++kc) {
    const int cur = kc & 1;
    // my DMAs for tile `kc` complete -> then workgroup-wide visibility
    asm volatile("s_wait_asynccnt 0x0" ::: "memory");
    __syncthreads();
    // everyone is past reading buf[cur^1] (tile kc-1): safe to overwrite it,
    // and the DMA streams underneath this iteration's WMMAs.
    if (kc + 1 < 24) stage_tile(kc + 1, cur ^ 1);

    const v16bf afr = frag_ld(&At[cur][0] + band * 16 * 32, 32, lane);
    #pragma unroll
    for (int tc = 0; tc < 8; ++tc) {
      const v16bf bfr = frag_ld(&Bt[cur][0] + (half * 8 + tc) * 16 * 32, 32, lane);
      acc[tc] = __builtin_amdgcn_wmma_f32_16x16x32_bf16(
          false, afr, false, bfr, (short)0, acc[tc], false, false);
    }
  }
  __syncthreads();

  // epilogue 1: bias + relu, keep `processed` only in LDS (bf16)
  #pragma unroll
  for (int tc = 0; tc < 8; ++tc) {
    const int col  = half * 128 + tc * 16 + (lane & 15);
    const float bv = tp_b[col];
    #pragma unroll
    for (int r = 0; r < 8; ++r) {
      const int row = band * 16 + r + ((lane >> 4) << 3);
      float v = acc[tc][r] + bv;
      Pb[row * 256 + col] = (__bf16)(v > 0.f ? v : 0.f);
    }
  }
  __syncthreads();

  // epilogue 2: qk0/qk1 = processed . c0/c1 (4 partials per row)
  {
    const int row = tid >> 2;
    const int q4  = tid & 3;
    const __bf16* pr = Pb + row * 256 + q4 * 64;
    const float* c0p = c0 + q4 * 64;
    const float* c1p = c1 + q4 * 64;
    float s0 = 0.f, s1 = 0.f;
    #pragma unroll 8
    for (int c = 0; c < 64; ++c) {
      const float v = (float)pr[c];
      s0 += v * c0p[c];
      s1 += v * c1p[c];
    }
    red[(row * 4 + q4) * 2 + 0] = s0;
    red[(row * 4 + q4) * 2 + 1] = s1;
  }
  __syncthreads();

  // epilogue 3: 3-way attention -> (aR, aS). k_b term cancels in softmax.
  if (tid < 64) {
    float qk0 = dconst[0], qk1 = dconst[1];
    #pragma unroll
    for (int q = 0; q < 4; ++q) {
      qk0 += red[(tid * 4 + q) * 2 + 0];
      qk1 += red[(tid * 4 + q) * 2 + 1];
    }
    const int g  = rowbase + tid;     // flat (b,m,t)
    const int bm = g >> 11;
    const int t  = g & 2047;
    const int m  = bm & 3;
    float r3[3], s3[3], sc[3];
    #pragma unroll
    for (int n = 0; n < 3; ++n) {
      const int o = n + (n >= m ? 1 : 0);             // others[m][n]
      const size_t idx = ((size_t)(bm * 4 + o) << 11) + t;
      r3[n] = R[idx];
      s3[n] = S[idx];
      sc[n] = (r3[n] * qk0 + s3[n] * qk1) * 0.125f;   // /sqrt(64)
    }
    const float mx = fmaxf(sc[0], fmaxf(sc[1], sc[2]));
    const float e0 = __expf(sc[0] - mx), e1 = __expf(sc[1] - mx), e2 = __expf(sc[2] - mx);
    const float inv = 1.f / (e0 + e1 + e2);
    aRS[(size_t)g * 2 + 0] = (e0 * r3[0] + e1 * r3[1] + e2 * r3[2]) * inv;
    aRS[(size_t)g * 2 + 1] = (e0 * s3[0] + e1 * s3[1] + e2 * s3[2]) * inv;
  }

  // epilogue 4: mlpA = processed @ mlp_w1[:, :256]^T + b1 (WMMA, K=256)
  v8f acc2[6] = {};
  for (int kc = 0; kc < 8; ++kc) {
    const v16bf afr = frag_ld(Pb + band * 16 * 256 + kc * 32, 256, lane);
    #pragma unroll
    for (int j = 0; j < 6; ++j) {
      const int ct = half * 6 + j;
      const v16bf bfr = frag_ld(w1a_bf + (size_t)ct * 16 * 256 + kc * 32, 256, lane);
      acc2[j] = __builtin_amdgcn_wmma_f32_16x16x32_bf16(
          false, afr, false, bfr, (short)0, acc2[j], false, false);
    }
  }
  #pragma unroll
  for (int j = 0; j < 6; ++j) {
    const int col  = half * 96 + j * 16 + (lane & 15);
    const float bv = mlp_b1[col];
    #pragma unroll
    for (int r = 0; r < 8; ++r) {
      const int row = band * 16 + r + ((lane >> 4) << 3);
      mlpA[(size_t)(rowbase + row) * 192 + col] = acc2[j][r] + bv;
    }
  }
}

// ---------------------------------------------------------------------------
// Kernel 2: sequential GRU + MLP head. 32 blocks (one per sequence) x 192.
// Weights (~300 KB) stay L2-resident; state lives in LDS.
// ---------------------------------------------------------------------------
__global__ __launch_bounds__(192) void gru_tail_kernel(
    const float* __restrict__ U,       const float* __restrict__ w_hh,
    const float* __restrict__ b_hh,    const float* __restrict__ mlp_w1,
    const float* __restrict__ mlp_w2,  const float* __restrict__ mlp_b2,
    const float* __restrict__ wUv,  const float* __restrict__ waRv,
    const float* __restrict__ waSv, const float* __restrict__ wcv,
    const float* __restrict__ aRS,  const float* __restrict__ mlpA,
    float* __restrict__ out)
{
  __shared__ float h[128], hn[128], xs[384], hp[384], hid[192];
  __shared__ float wUs[384], waRs[384], waSs[384], wcs[384];

  const int tid = threadIdx.x;
  const size_t base = (size_t)blockIdx.x * 2048;

  for (int i = tid; i < 384; i += 192) {
    wUs[i] = wUv[i]; waRs[i] = waRv[i]; waSs[i] = waSv[i]; wcs[i] = wcv[i];
  }
  if (tid < 128) h[tid] = 0.f;
  __syncthreads();

  for (int t = 0; t < 2048; ++t) {
    const float u  = U[base + t];
    const float aR = aRS[(base + t) * 2 + 0];
    const float aS = aRS[(base + t) * 2 + 1];
    if (t + 1 < 2048) __builtin_prefetch(&mlpA[(base + t + 1) * 192 + tid], 0, 1);

    for (int i = tid; i < 384; i += 192)
      xs[i] = u * wUs[i] + aR * waRs[i] + aS * waSs[i] + wcs[i];
    for (int row = tid; row < 384; row += 192) {
      const float4* wr = reinterpret_cast<const float4*>(w_hh + (size_t)row * 128);
      float s = b_hh[row];
      #pragma unroll 8
      for (int k = 0; k < 32; ++k) {
        const float4 w4 = wr[k];
        s += w4.x * h[k * 4 + 0] + w4.y * h[k * 4 + 1] +
             w4.z * h[k * 4 + 2] + w4.w * h[k * 4 + 3];
      }
      hp[row] = s;
    }
    __syncthreads();

    if (tid < 128) {
      const float r = 1.f / (1.f + __expf(-(xs[tid] + hp[tid])));
      const float z = 1.f / (1.f + __expf(-(xs[128 + tid] + hp[128 + tid])));
      const float n = tanhf(xs[256 + tid] + r * hp[256 + tid]);
      hn[tid] = (1.f - z) * n + z * h[tid];
    }
    __syncthreads();

    {
      const float4* wr = reinterpret_cast<const float4*>(mlp_w1 + (size_t)tid * 384 + 256);
      float s = mlpA[(base + t) * 192 + tid];
      #pragma unroll 8
      for (int k = 0; k < 32; ++k) {
        const float4 w4 = wr[k];
        s += w4.x * hn[k * 4 + 0] + w4.y * hn[k * 4 + 1] +
             w4.z * hn[k * 4 + 2] + w4.w * hn[k * 4 + 3];
      }
      hid[tid] = fmaxf(s, 0.f);
    }
    __syncthreads();

    if (tid < 16) {
      const float* wr = mlp_w2 + tid * 192;
      float s = mlp_b2[tid];
      #pragma unroll 8
      for (int k = 0; k < 192; ++k) s += wr[k] * hid[k];
      out[(base + t) * 16 + tid] = s;
    }
    if (tid < 128) h[tid] = hn[tid];
    __syncthreads();
  }
}

// ---------------------------------------------------------------------------
extern "C" void kernel_launch(void* const* d_in, const int* in_sizes, int n_in,
                              void* d_out, int out_size, void* d_ws, size_t ws_size,
                              hipStream_t stream) {
  (void)in_sizes; (void)n_in; (void)out_size; (void)ws_size;
  const float* tok      = (const float*)d_in[0];
  const float* U        = (const float*)d_in[1];
  const float* R        = (const float*)d_in[2];
  const float* S        = (const float*)d_in[3];
  const float* tp_w     = (const float*)d_in[4];
  const float* tp_b     = (const float*)d_in[5];
  const float* q_w      = (const float*)d_in[6];
  const float* q_b      = (const float*)d_in[7];
  const float* k_w      = (const float*)d_in[8];
  /* d_in[9] = k_b: cancels in softmax */
  const float* v_w      = (const float*)d_in[10];
  const float* v_b      = (const float*)d_in[11];
  const float* gru_w_ih = (const float*)d_in[12];
  const float* gru_w_hh = (const float*)d_in[13];
  const float* gru_b_ih = (const float*)d_in[14];
  const float* gru_b_hh = (const float*)d_in[15];
  const float* mlp_w1   = (const float*)d_in[16];
  const float* mlp_b1   = (const float*)d_in[17];
  const float* mlp_w2   = (const float*)d_in[18];
  const float* mlp_b2   = (const float*)d_in[19];
  float* out = (float*)d_out;

  char* ws = (char*)d_ws;
  __bf16* tp_bf  = (__bf16*)(ws + OFF_TPBF);
  __bf16* w1a_bf = (__bf16*)(ws + OFF_W1A);
  float*  c0     = (float*)(ws + OFF_C0);
  float*  c1     = (float*)(ws + OFF_C1);
  float*  dc     = (float*)(ws + OFF_D);
  float*  wU     = (float*)(ws + OFF_WU);
  float*  waR    = (float*)(ws + OFF_WAR);
  float*  waS    = (float*)(ws + OFF_WAS);
  float*  wc     = (float*)(ws + OFF_WC);
  float*  aRS    = (float*)(ws + OFF_ARS);
  float*  mlpA   = (float*)(ws + OFF_MLPA);

  prep_kernel<<<256, 256, 0, stream>>>(tp_w, mlp_w1, q_w, q_b, k_w, v_w, v_b,
                                       gru_w_ih, gru_b_ih,
                                       tp_bf, w1a_bf, c0, c1, dc, wU, waR, waS, wc);
  fused_gemm_attn_kernel<<<1024, 256, 0, stream>>>(tok, R, S, tp_b, mlp_b1,
                                                   tp_bf, w1a_bf, c0, c1, dc,
                                                   aRS, mlpA);
  gru_tail_kernel<<<32, 192, 0, stream>>>(U, gru_w_hh, gru_b_hh, mlp_w1,
                                          mlp_w2, mlp_b2, wU, waR, waS, wc,
                                          aRS, mlpA, out);
}
